// QWenAttention_1726576854701
// MI455X (gfx1250) — matrix operand order repaired and checked by reading, not compiled, not run on hardware
//
#include <hip/hip_runtime.h>
#include <hip/hip_bf16.h>

// ---------------- problem constants ----------------
constexpr int B_  = 2;
constexpr int S_  = 2048;
constexpr int H_  = 4096;
constexpr int NH_ = 32;
constexpr int HD_ = 128;
constexpr float SM_SCALE = 0.08838834764831845f;   // 1/sqrt(128)
constexpr float LOG2E    = 1.4426950408889634f;

// ---------------- types ----------------
typedef __bf16 bf16;
typedef bf16  bf16x8  __attribute__((ext_vector_type(8)));
typedef bf16  bf16x16 __attribute__((ext_vector_type(16)));
typedef float f32x8   __attribute__((ext_vector_type(8)));

union BF16x16 { bf16x16 v; bf16x8 h[2]; };

// ---------------- fp32 -> bf16 convert (vector x4) ----------------
__global__ __launch_bounds__(256) void cvt_f32_bf16(const float* __restrict__ in,
                                                    bf16* __restrict__ out, size_t n4) {
  size_t i = (size_t)blockIdx.x * blockDim.x + threadIdx.x;
  if (i >= n4) return;
  const float4 f = ((const float4*)in)[i];
  bf16* o = out + i * 4;
  o[0] = (bf16)f.x; o[1] = (bf16)f.y; o[2] = (bf16)f.z; o[3] = (bf16)f.w;
}

// ---------------- GEMM: C[M,N] = A[M,K](bf16) * W[N,K](bf16)^T + bias ----------------
// one wave -> 32x64 tile (2 M-subtiles x 4 N-subtiles = 8 accumulators);
// each B (weight) tile is reused by two WMMAs -> halves weight-stream VMEM per FLOP.
__global__ __launch_bounds__(256) void gemm_bf16_nt(const bf16* __restrict__ A,
                                                    const bf16* __restrict__ W,
                                                    const float* __restrict__ bias,
                                                    float* __restrict__ C,
                                                    int M, int N, int K) {
  const int lane = threadIdx.x & 31;
  const int wave = threadIdx.x >> 5;
  const int ntiles4 = N >> 6;            // 64-wide N tiles
  const int mtiles  = M >> 5;            // 32-tall M tiles
  const int flat = blockIdx.x * 8 + wave;
  if (flat >= mtiles * ntiles4) return;
  const int mt = flat / ntiles4;
  const int nt = flat % ntiles4;

  const int r16  = lane & 15;
  const int half = lane >> 4;
  const int csA  = half * 8;             // A-layout K split {0..7,16..23}/{8..15,24..31}
  const int csB  = half * 16;            // cancelled in B loads (wp - csB); B uses the same K split as A

  const bf16* arow0 = A + (size_t)(mt * 32 + r16) * K;
  const bf16* arow1 = arow0 + (size_t)16 * K;
  const bf16* wrow  = W + (size_t)(nt * 64 + r16) * K;

  f32x8 acc[2][4];
#pragma unroll
  for (int m = 0; m < 2; ++m)
#pragma unroll
    for (int t = 0; t < 4; ++t) acc[m][t] = f32x8{};

  for (int kk = 0; kk < K; kk += 32) {
    BF16x16 a0, a1;
    a0.h[0] = *(const bf16x8*)(arow0 + kk + csA);
    a0.h[1] = *(const bf16x8*)(arow0 + kk + 16 + csA);
    a1.h[0] = *(const bf16x8*)(arow1 + kk + csA);
    a1.h[1] = *(const bf16x8*)(arow1 + kk + 16 + csA);
#pragma unroll
    for (int t = 0; t < 4; ++t) {
      const bf16* wp = wrow + (size_t)(t * 16) * K + kk + csB;
      BF16x16 b;
      b.h[0] = *(const bf16x8*)(wp - csB + csA);
      b.h[1] = *(const bf16x8*)(wp - csB + 16 + csA);
      acc[0][t] = __builtin_amdgcn_wmma_f32_16x16x32_bf16(false, a0.v, false, b.v,
                                                          (short)0, acc[0][t], false, false);
      acc[1][t] = __builtin_amdgcn_wmma_f32_16x16x32_bf16(false, a1.v, false, b.v,
                                                          (short)0, acc[1][t], false, false);
    }
  }
  // C-layout: lane -> col = lane&15 ; vgpr r -> row = r + 8*(lane>>4)
#pragma unroll
  for (int t = 0; t < 4; ++t) {
    const int col = nt * 64 + t * 16 + r16;
    const float bv = bias ? bias[col] : 0.0f;
#pragma unroll
    for (int m = 0; m < 2; ++m) {
#pragma unroll
      for (int r = 0; r < 8; ++r) {
        const int orow = mt * 32 + m * 16 + r + half * 8;
        C[(size_t)orow * N + col] = acc[m][t][r] + bv;
      }
    }
  }
}

// ---------------- RoPE + logn, emit bf16 Q/K [B,NH,S,HD] and V^T [B,NH,HD,S] ----------
__global__ __launch_bounds__(128) void rope_logn_kernel(const float* __restrict__ qkv,
                                                        const float* __restrict__ cosb,
                                                        const float* __restrict__ sinb,
                                                        const float* __restrict__ logn,
                                                        bf16* __restrict__ qb,
                                                        bf16* __restrict__ kb,
                                                        bf16* __restrict__ vT) {
  const int d  = threadIdx.x;          // 0..127
  const int h  = blockIdx.x % NH_;
  const int bs = blockIdx.x / NH_;     // b*S + s
  const int s  = bs % S_;
  const int b  = bs / S_;

  const size_t rowoff = (size_t)bs * (3 * H_);
  const float q  = qkv[rowoff +           h * HD_ + d];
  const float k  = qkv[rowoff +     H_  + h * HD_ + d];
  const float v  = qkv[rowoff + 2 * H_  + h * HD_ + d];
  const int   dp = (d < 64) ? d + 64 : d - 64;
  const float qp = qkv[rowoff +           h * HD_ + dp];
  const float kp = qkv[rowoff +     H_  + h * HD_ + dp];

  const float c  = cosb[s * HD_ + d];
  const float sn = sinb[s * HD_ + d];
  const float rq = (d < 64) ? -qp : qp;
  const float rk = (d < 64) ? -kp : kp;
  const float qo = (q * c + rq * sn) * logn[s];
  const float ko =  k * c + rk * sn;

  const size_t bh = (size_t)b * NH_ + h;
  qb[(bh * S_ + s) * HD_ + d] = (bf16)qo;
  kb[(bh * S_ + s) * HD_ + d] = (bf16)ko;
  vT[(bh * HD_ + d) * S_ + s] = (bf16)v;
}

// ---------------- flash attention: one wave per (b,h, 16 q-rows) -----------------------
__global__ __launch_bounds__(256) void flash_attn_kernel(const bf16* __restrict__ qb,
                                                         const bf16* __restrict__ kb,
                                                         const bf16* __restrict__ vT,
                                                         bf16* __restrict__ ctx) {
  __shared__ float lds[8][16][33];               // per-wave probs staging (+pad)
  const int lane = threadIdx.x & 31;
  const int wave = threadIdx.x >> 5;
  const int qtiles = S_ / 16;
  const int job = blockIdx.x * 8 + wave;
  const int bh  = job / qtiles;
  const int qt  = job % qtiles;
  if (bh >= B_ * NH_) return;

  const int qbase = qt * 16;
  const int r16   = lane & 15;
  const int half  = lane >> 4;
  const int csA   = half * 8;
  const int csB   = half * 16;

  // Q tile in WMMA A-layout: lane r16 holds q-row (qbase+r16)
  const bf16* qrow = qb + ((size_t)bh * S_ + qbase + r16) * HD_;
  BF16x16 aq[4];
#pragma unroll
  for (int t = 0; t < 4; ++t) {
    aq[t].h[0] = *(const bf16x8*)(qrow + t * 32 + csA);
    aq[t].h[1] = *(const bf16x8*)(qrow + t * 32 + 16 + csA);
  }

  f32x8 acc[8];
#pragma unroll
  for (int dt = 0; dt < 8; ++dt) acc[dt] = f32x8{};
  float mrow[8], lrow[8];
#pragma unroll
  for (int r = 0; r < 8; ++r) { mrow[r] = -1e30f; lrow[r] = 0.0f; }

  const int nblk = (qbase + 16 + 31) >> 5;       // causal: keys <= qbase+15
  for (int j = 0; j < nblk; ++j) {
    const int kbase = j * 32;

    // ---- scores: S(16x32) = Q(16x128) * K^T ; two 16x16 WMMA chains over K=128 ----
    const bf16* krow0 = kb + ((size_t)bh * S_ + kbase + r16) * HD_;
    const bf16* krow1 = krow0 + (size_t)16 * HD_;
    f32x8 s0 = f32x8{}, s1 = f32x8{};
#pragma unroll
    for (int t = 0; t < 4; ++t) {
      BF16x16 b0, b1;
      b0.h[0] = *(const bf16x8*)(krow0 + t * 32 + csA);
      b0.h[1] = *(const bf16x8*)(krow0 + t * 32 + 16 + csA);
      b1.h[0] = *(const bf16x8*)(krow1 + t * 32 + csA);
      b1.h[1] = *(const bf16x8*)(krow1 + t * 32 + 16 + csA);
      s0 = __builtin_amdgcn_wmma_f32_16x16x32_bf16(false, aq[t].v, false, b0.v,
                                                   (short)0, s0, false, false);
      s1 = __builtin_amdgcn_wmma_f32_16x16x32_bf16(false, aq[t].v, false, b1.v,
                                                   (short)0, s1, false, false);
    }

    // ---- online softmax over the 16x32 block ----
    float alpha[8];
#pragma unroll
    for (int r = 0; r < 8; ++r) {
      const int qr = qbase + r + half * 8;
      float v0 = s0[r] * SM_SCALE;
      float v1 = s1[r] * SM_SCALE;
      if (kbase +      r16 > qr) v0 = -1e30f;
      if (kbase + 16 + r16 > qr) v1 = -1e30f;
      float mx = fmaxf(v0, v1);
      mx = fmaxf(mx, __shfl_xor(mx, 1, 32));
      mx = fmaxf(mx, __shfl_xor(mx, 2, 32));
      mx = fmaxf(mx, __shfl_xor(mx, 4, 32));
      mx = fmaxf(mx, __shfl_xor(mx, 8, 32));
      const float mnew = fmaxf(mrow[r], mx);
      const float p0 = exp2f((v0 - mnew) * LOG2E);
      const float p1 = exp2f((v1 - mnew) * LOG2E);
      const float a  = exp2f((mrow[r] - mnew) * LOG2E);
      float ps = p0 + p1;
      ps += __shfl_xor(ps, 1, 32);
      ps += __shfl_xor(ps, 2, 32);
      ps += __shfl_xor(ps, 4, 32);
      ps += __shfl_xor(ps, 8, 32);
      lrow[r] = lrow[r] * a + ps;
      mrow[r] = mnew;
      alpha[r] = a;
      lds[wave][r + half * 8][r16]      = p0;    // row-major probs for relayout
      lds[wave][r + half * 8][16 + r16] = p1;
    }
    // rescale accumulators by alpha (per row)
#pragma unroll
    for (int dt = 0; dt < 8; ++dt)
#pragma unroll
      for (int r = 0; r < 8; ++r) acc[dt][r] *= alpha[r];

    asm volatile("" ::: "memory");               // keep DS store->load ordered (in-order per wave)

    // re-load probs in WMMA A-layout, convert to bf16
    BF16x16 pa;
#pragma unroll
    for (int i = 0; i < 8; ++i) {
      pa.v[i]     = (bf16)lds[wave][r16][csA + i];
      pa.v[8 + i] = (bf16)lds[wave][r16][16 + csA + i];
    }

    // ---- ctx += P(16x32) * V(32x128) ; V^T rows are contiguous keys ----
#pragma unroll
    for (int dt = 0; dt < 8; ++dt) {
      const bf16* vrow = vT + ((size_t)bh * HD_ + dt * 16 + r16) * S_ + kbase;
      BF16x16 bv;
      bv.h[0] = *(const bf16x8*)(vrow + csA);
      bv.h[1] = *(const bf16x8*)(vrow + 16 + csA);
      acc[dt] = __builtin_amdgcn_wmma_f32_16x16x32_bf16(false, pa.v, false, bv.v,
                                                        (short)0, acc[dt], false, false);
    }
  }

  // ---- epilogue: normalize, store bf16 ctx [B,S,H] ----
  const int b = bh / NH_;
  const int h = bh % NH_;
#pragma unroll
  for (int dt = 0; dt < 8; ++dt) {
#pragma unroll
    for (int r = 0; r < 8; ++r) {
      const int srow = qbase + r + half * 8;
      const float o  = acc[dt][r] / lrow[r];
      ctx[((size_t)b * S_ + srow) * H_ + h * HD_ + dt * 16 + r16] = (bf16)o;
    }
  }
}

// ---------------- host side ----------------
extern "C" void kernel_launch(void* const* d_in, const int* in_sizes, int n_in,
                              void* d_out, int out_size, void* d_ws, size_t ws_size,
                              hipStream_t stream) {
  const float* hidden   = (const float*)d_in[0];
  const float* w_attn   = (const float*)d_in[1];
  const float* b_attn   = (const float*)d_in[2];
  const float* w_proj   = (const float*)d_in[3];
  const float* b_proj   = (const float*)d_in[4];
  const float* rope_cos = (const float*)d_in[5];
  const float* rope_sin = (const float*)d_in[6];
  const float* logn     = (const float*)d_in[7];
  float* out = (float*)d_out;

  char* ws = (char*)d_ws;
  size_t off = 0;
  auto carve = [&](size_t bytes) -> char* {
    char* p = ws + off;
    off += (bytes + 255) & ~(size_t)255;
    return p;
  };

  const size_t nHid  = (size_t)B_ * S_ * H_;          // 16.7M
  const size_t nWat  = (size_t)3 * H_ * H_;           // 50.3M
  const size_t nWpj  = (size_t)H_ * H_;               // 16.7M
  const size_t nQKV  = (size_t)B_ * S_ * 3 * H_;      // 50.3M
  const size_t nHead = (size_t)B_ * NH_ * S_ * HD_;   // 16.7M

  bf16*  hiddenB = (bf16*)carve(nHid  * sizeof(bf16));
  bf16*  wAttnB  = (bf16*)carve(nWat  * sizeof(bf16));
  bf16*  wProjB  = (bf16*)carve(nWpj  * sizeof(bf16));
  float* qkv     = (float*)carve(nQKV * sizeof(float));
  bf16*  qB      = (bf16*)carve(nHead * sizeof(bf16));
  bf16*  kB      = (bf16*)carve(nHead * sizeof(bf16));
  bf16*  vT      = (bf16*)carve(nHead * sizeof(bf16));
  bf16*  ctxB    = (bf16*)carve(nHid  * sizeof(bf16));

  // 1) converts
  {
    size_t n4 = nHid / 4;
    cvt_f32_bf16<<<dim3((unsigned)((n4 + 255) / 256)), dim3(256), 0, stream>>>(hidden, hiddenB, n4);
    n4 = nWat / 4;
    cvt_f32_bf16<<<dim3((unsigned)((n4 + 255) / 256)), dim3(256), 0, stream>>>(w_attn, wAttnB, n4);
    n4 = nWpj / 4;
    cvt_f32_bf16<<<dim3((unsigned)((n4 + 255) / 256)), dim3(256), 0, stream>>>(w_proj, wProjB, n4);
  }

  // 2) QKV projection: [4096,12288] = hidden[4096,4096] * w_attn^T + b_attn
  {
    const int M = B_ * S_, N = 3 * H_, K = H_;
    const int tiles = (M / 32) * (N / 64);
    gemm_bf16_nt<<<dim3((tiles + 7) / 8), dim3(256), 0, stream>>>(hiddenB, wAttnB, b_attn, qkv, M, N, K);
  }

  // 3) RoPE + logn + layout
  rope_logn_kernel<<<dim3(B_ * S_ * NH_), dim3(128), 0, stream>>>(qkv, rope_cos, rope_sin, logn,
                                                                  qB, kB, vT);

  // 4) causal flash attention
  {
    const int jobs = B_ * NH_ * (S_ / 16);   // 8192 waves
    flash_attn_kernel<<<dim3(jobs / 8), dim3(256), 0, stream>>>(qB, kB, vT, ctxB);
  }

  // 5) output projection: [4096,4096] = ctx * w_proj^T + b_proj
  {
    const int M = B_ * S_, N = H_, K = H_;
    const int tiles = (M / 32) * (N / 64);
    gemm_bf16_nt<<<dim3((tiles + 7) / 8), dim3(256), 0, stream>>>(ctxB, wProjB, b_proj, out, M, N, K);
  }
}
